// MultHeadAttention_43903155700389
// MI455X (gfx1250) — compile-verified, matrix operand order checked by reading, not run on hardware
//
#include <hip/hip_runtime.h>

#define DEVINL __device__ __forceinline__

typedef __attribute__((ext_vector_type(16))) _Float16 v16h;
typedef __attribute__((ext_vector_type(8)))  _Float16 v8h;
typedef __attribute__((ext_vector_type(4)))  _Float16 v4h;
typedef __attribute__((ext_vector_type(8)))  float    v8f;
typedef __attribute__((ext_vector_type(4)))  float    v4f;

constexpr int B_  = 4096;
constexpr int S_  = 50;
constexpr int D_  = 32;
constexpr int H_  = 4;
constexpr int SP  = 64;                 // padded sequence length (4 x 16 row tiles)
constexpr float SCALE = 0.35355339059327373f;  // 1/sqrt(8)

// ---- LDS layout (dynamic shared) -------------------------------------------
// [0,8192)     : f16 weights TRANSPOSED  WQt|WKt|WVt|WOt, each [32 n][32 k]
// [8192,8704)  : f32 biases  bQ|bK|bV|bO, each [32]
// per wave w (base = 8704 + w*27648):
//   +0     : t0    [64][32] f16   (aliased by attn)
//   +4096  : t1    [64][32] f16   (aliased by attn)
//   +0     : attn  [64][64] f16   (reuses t0+t1 after linears are done)
//   +8192  : qf    [64][32] f16   (row-major, feeds A-frags)
//   +12288 : kf    [64][32] f16   (row-major, feeds score B-frags = row loads)
//   +16384 : vt    [32][64] f16   (TRANSPOSED v, feeds ctx B-frags = row loads)
//   +20480 : ctxb  [64][32] f16
//   +24576 : smask [50*50] u8 (row stride 50, raw copy of the mask)
constexpr int WAVE_LDS   = 27648;
constexpr int SMEM_BYTES = 8704 + 4 * WAVE_LDS;   // 119296 B

DEVINL v8f wmma_f32_f16(v16h a, v16h b, v8f c) {
    return __builtin_amdgcn_wmma_f32_16x16x32_f16(
        /*neg_a=*/false, a, /*neg_b=*/false, b,
        /*c_mod=*/(short)0, c, /*reuse_a=*/false, /*reuse_b=*/false);
}

// A-style fragment load (16 rows x 32 k, f16): lane holds row rowBase+lane%16,
// two contiguous 16B chunks select the lane-half's K slots. Used for:
//   - A of every GEMM (row-major tiles)
//   - B of q*k^T        (B[k][n] = K[n][k]  -> row load of row-major K)
//   - B of X@W          (B[k][n] = Wt[n][k] -> row load of transposed W)
//   - B of attn@v       (B[k][n] = vt[n'][k] -> row load of transposed V)
DEVINL v16h load_frag(const _Float16* buf, int rowBase, int rowStride, int kOfs, int lane) {
    int row = rowBase + (lane & 15);
    const _Float16* p = buf + row * rowStride + kOfs + ((lane & 16) ? 8 : 0);
    v8h lo = *(const v8h*)(p);
    v8h hi = *(const v8h*)(p + 16);
    return __builtin_shufflevector(lo, hi, 0,1,2,3,4,5,6,7,8,9,10,11,12,13,14,15);
}

// Store C/D fragment (16x16 f32) as f16, row-major tile.
DEVINL void store_c_f16(_Float16* buf, int rowStride, int rowBase, int colBase,
                        const v8f& c, int lane) {
    int col = colBase + (lane & 15);
    int r0  = rowBase + ((lane & 16) ? 8 : 0);
#pragma unroll
    for (int r = 0; r < 8; ++r)
        buf[(r0 + r) * rowStride + col] = (_Float16)c[r];
}

// Store C/D fragment TRANSPOSED: bufT[col][row]; the 8 regs are 8 consecutive
// rows -> one contiguous 16B store per lane.
DEVINL void store_c_f16_T(_Float16* bufT, int rowStrideT, int rowBase, int colBase,
                          const v8f& c, int lane) {
    int col = colBase + (lane & 15);
    int r0  = rowBase + ((lane & 16) ? 8 : 0);
    v8h h;
#pragma unroll
    for (int r = 0; r < 8; ++r) h[r] = (_Float16)c[r];
    *(v8h*)(bufT + col * rowStrideT + r0) = h;
}

// dst = src[64][32] @ W + bias, W given transposed ([32 n][32 k]).
// TRANS=false: dst row-major [64][32]; TRANS=true: dst transposed [32][64].
template <bool TRANS>
DEVINL void linear32(const _Float16* src, const _Float16* Wt, const float* bias,
                     _Float16* dst, int lane) {
    v16h bw[2];
#pragma unroll
    for (int nt = 0; nt < 2; ++nt)
        bw[nt] = load_frag(Wt, nt * 16, 32, 0, lane);
#pragma unroll
    for (int mt = 0; mt < 4; ++mt) {
        v16h a = load_frag(src, mt * 16, 32, 0, lane);
#pragma unroll
        for (int nt = 0; nt < 2; ++nt) {
            v8f c = {};
            c = wmma_f32_f16(a, bw[nt], c);
            float bv = bias[nt * 16 + (lane & 15)];
#pragma unroll
            for (int r = 0; r < 8; ++r) c[r] += bv;
            if (TRANS) store_c_f16_T(dst, 64, mt * 16, nt * 16, c, lane);
            else       store_c_f16 (dst, 32, mt * 16, nt * 16, c, lane);
        }
    }
}

// Load one [50][32] f32 matrix -> f16 LDS [64][32], zero-padding rows 50..63.
// float4 global loads, 8B LDS stores.
DEVINL void load_x_f16(const float* __restrict__ g, _Float16* dst, int lane) {
#pragma unroll 4
    for (int i = lane; i < (SP * 32) / 4; i += 32) {      // 512 float4 groups
        v4f x = {0.f, 0.f, 0.f, 0.f};
        if (i < (S_ * 32) / 4) x = ((const v4f*)g)[i];    // 400 real groups
        v4h h;
#pragma unroll
        for (int e = 0; e < 4; ++e) h[e] = (_Float16)x[e];
        *(v4h*)(dst + i * 4) = h;
    }
}

__global__ __launch_bounds__(128)
void mha_wmma_kernel(const float* __restrict__ gQ, const float* __restrict__ gK,
                     const float* __restrict__ gV, const unsigned char* __restrict__ gMask,
                     const float* __restrict__ gWQ, const float* __restrict__ gbQ,
                     const float* __restrict__ gWK, const float* __restrict__ gbK,
                     const float* __restrict__ gWV, const float* __restrict__ gbV,
                     const float* __restrict__ gWO, const float* __restrict__ gbO,
                     float* __restrict__ gOut) {
    extern __shared__ char smem[];
    _Float16* sWt = (_Float16*)smem;            // 4 x [32 n][32 k] (transposed)
    float*    sB  = (float*)(smem + 8192);      // 4 x [32]

    const int tid  = threadIdx.x;
    const int lane = tid & 31;
    const int wave = tid >> 5;
    const int half = (lane >> 4) & 1;
    const int lcol = lane & 15;
    const int b    = blockIdx.x * 4 + wave;

    // --- weights: f32 global -> f16 LDS, TRANSPOSED (one-time, per block) ---
    for (int i = tid; i < 1024; i += 128) {
        int k = i >> 5, n = i & 31;
        int t = n * 32 + k;
        sWt[t]        = (_Float16)gWQ[i];
        sWt[1024 + t] = (_Float16)gWK[i];
        sWt[2048 + t] = (_Float16)gWV[i];
        sWt[3072 + t] = (_Float16)gWO[i];
    }
    if (tid < 32) {
        sB[tid]      = gbQ[tid];
        sB[32 + tid] = gbK[tid];
        sB[64 + tid] = gbV[tid];
        sB[96 + tid] = gbO[tid];
    }
    __syncthreads();

    char* wb = smem + 8704 + wave * WAVE_LDS;
    _Float16* t0   = (_Float16*)(wb);
    _Float16* t1   = (_Float16*)(wb + 4096);
    _Float16* attn = (_Float16*)(wb);           // aliases t0/t1 (used later)
    _Float16* qf   = (_Float16*)(wb + 8192);
    _Float16* kf   = (_Float16*)(wb + 12288);
    _Float16* vt   = (_Float16*)(wb + 16384);   // [32 d][64 s]
    _Float16* ctxb = (_Float16*)(wb + 20480);
    unsigned char* smask = (unsigned char*)(wb + 24576);  // [50][50], stride 50

    const size_t xoff = (size_t)b * (S_ * D_);

    // ---------------- double-applied QKV linears (f16 WMMA) -----------------
    __builtin_prefetch(gK + xoff, 0, 1);        // global_prefetch_b8
    load_x_f16(gQ + xoff, t0, lane);
    linear32<false>(t0, sWt + 0,    sB + 0,  t1, lane);
    linear32<false>(t1, sWt + 0,    sB + 0,  qf, lane);

    __builtin_prefetch(gV + xoff, 0, 1);
    load_x_f16(gK + xoff, t0, lane);
    linear32<false>(t0, sWt + 1024, sB + 32, t1, lane);
    linear32<false>(t1, sWt + 1024, sB + 32, kf, lane);

    __builtin_prefetch(gMask + (size_t)b * (S_ * S_), 0, 1);
    load_x_f16(gV + xoff, t0, lane);
    linear32<false>(t0, sWt + 2048, sB + 64, t1, lane);
    linear32<true >(t1, sWt + 2048, sB + 64, vt, lane);   // -> transposed vt

    // ---------------- mask -> LDS (raw u32 copy, stride 50) -----------------
    {
        const unsigned int* gm32 = (const unsigned int*)(gMask + (size_t)b * (S_ * S_));
        unsigned int* sm32 = (unsigned int*)smask;
        for (int i = lane; i < (S_ * S_) / 4; i += 32)    // 2500/4 = 625 words
            sm32[i] = gm32[i];
    }

    // ---------------- attention -------------------------------------------
    v8f zero = {};
    v8f ctxAcc[4][2];
#pragma unroll
    for (int mt = 0; mt < 4; ++mt)
#pragma unroll
        for (int nt = 0; nt < 2; ++nt) ctxAcc[mt][nt] = zero;

    // head-invariant fragments: q rows (A) and k rows (B of q*k^T)
    v16h qA[4], kB[4];
#pragma unroll
    for (int t = 0; t < 4; ++t) {
        qA[t] = load_frag(qf, t * 16, 32, 0, lane);
        kB[t] = load_frag(kf, t * 16, 32, 0, lane);
    }

    // head-invariant mask bits: per row-tile mt, this lane touches 8 rows x
    // 4 col-tiles -> pack into one u32 (bit r*4+nt), padding rows/cols masked.
    unsigned int mbits[4];
#pragma unroll
    for (int mt = 0; mt < 4; ++mt) {
        unsigned int m = 0;
#pragma unroll
        for (int r = 0; r < 8; ++r) {
            int row = mt * 16 + half * 8 + r;
#pragma unroll
            for (int nt = 0; nt < 4; ++nt) {
                int col = nt * 16 + lcol;
                unsigned char mb = 1;
                if (row < S_ && col < S_) mb = smask[row * S_ + col];
                if (mb) m |= (1u << (r * 4 + nt));
            }
        }
        mbits[mt] = m;
    }

    const float NEG_INF = -__builtin_inff();

#pragma unroll
    for (int h = 0; h < H_; ++h) {
        // scores + softmax for this head, one 16-row tile at a time
#pragma unroll
        for (int mt = 0; mt < 4; ++mt) {
            // A = q rows with K slots outside this head's 8 dims zeroed.
            // head(j,half) = 2*(j>=4) + half, so the condition folds per group.
            v16h a = qA[mt];
#pragma unroll
            for (int j = 0; j < 8; ++j) {
                if ((2 * ((j >> 2) & 1) + half) != h) {
                    a[2 * j]     = (_Float16)0.f;
                    a[2 * j + 1] = (_Float16)0.f;
                }
            }
            v8f sc[4];
#pragma unroll
            for (int nt = 0; nt < 4; ++nt)
                sc[nt] = wmma_f32_f16(a, kB[nt], zero);

            // row-wise softmax on C-fragment layout
#pragma unroll
            for (int r = 0; r < 8; ++r) {
                int row = mt * 16 + half * 8 + r;
                float v[4];
                float mx = NEG_INF;
#pragma unroll
                for (int nt = 0; nt < 4; ++nt) {
                    float s = sc[nt][r] * SCALE;
                    if (mbits[mt] & (1u << (r * 4 + nt))) s = NEG_INF;
                    v[nt] = s;
                    mx = fmaxf(mx, s);
                }
#pragma unroll
                for (int off = 1; off < 16; off <<= 1)
                    mx = fmaxf(mx, __shfl_xor(mx, off, 32));
                float e[4];
                float sum = 0.f;
#pragma unroll
                for (int nt = 0; nt < 4; ++nt) {
                    e[nt] = __expf(v[nt] - mx);
                    sum += e[nt];
                }
#pragma unroll
                for (int off = 1; off < 16; off <<= 1)
                    sum += __shfl_xor(sum, off, 32);
                float inv = 1.f / sum;
#pragma unroll
                for (int nt = 0; nt < 4; ++nt)
                    attn[row * 64 + nt * 16 + lcol] = (_Float16)(e[nt] * inv);
            }
        }

        // ctx += attn_h @ v_h  (B from transposed vt = vector row loads;
        //  two heads share each 16-wide output tile, other head's cols zeroed)
        const int nt  = h >> 1;
        const int par = h & 1;
#pragma unroll
        for (int ks = 0; ks < 2; ++ks) {
            v16h bv = load_frag(vt, nt * 16, 64, ks * 32, lane);
            if ((lcol >> 3) != par) {
#pragma unroll
                for (int e = 0; e < 16; ++e) bv[e] = (_Float16)0.f;
            }
#pragma unroll
            for (int mt = 0; mt < 4; ++mt) {
                v16h a = load_frag(attn, mt * 16, 64, ks * 32, lane);
                ctxAcc[mt][nt] = wmma_f32_f16(a, bv, ctxAcc[mt][nt]);
            }
        }
    }

    // ctx -> f16 LDS for the output projection
#pragma unroll
    for (int mt = 0; mt < 4; ++mt)
#pragma unroll
        for (int nt = 0; nt < 2; ++nt)
            store_c_f16(ctxb, 32, mt * 16, nt * 16, ctxAcc[mt][nt], lane);

    // ---------------- out = ctx @ WO + bO + Qsrc ---------------------------
    const float* gq = gQ + xoff;
    float*       go = gOut + xoff;
    v16h bw[2];
#pragma unroll
    for (int nt = 0; nt < 2; ++nt)
        bw[nt] = load_frag(sWt + 3072, nt * 16, 32, 0, lane);
#pragma unroll
    for (int mt = 0; mt < 4; ++mt) {
        v16h a = load_frag(ctxb, mt * 16, 32, 0, lane);
#pragma unroll
        for (int nt = 0; nt < 2; ++nt) {
            v8f c = wmma_f32_f16(a, bw[nt], zero);
            float bvs = sB[96 + nt * 16 + lcol];
            int col = nt * 16 + lcol;
#pragma unroll
            for (int r = 0; r < 8; ++r) {
                int row = mt * 16 + half * 8 + r;
                if (row < S_)
                    go[row * 32 + col] = c[r] + bvs + gq[row * 32 + col];
            }
        }
    }
}

extern "C" void kernel_launch(void* const* d_in, const int* in_sizes, int n_in,
                              void* d_out, int out_size, void* d_ws, size_t ws_size,
                              hipStream_t stream) {
    const float*         gQ    = (const float*)d_in[0];
    const float*         gK    = (const float*)d_in[1];
    const float*         gV    = (const float*)d_in[2];
    const unsigned char* gMask = (const unsigned char*)d_in[3];
    const float*         gWQ   = (const float*)d_in[4];
    const float*         gbQ   = (const float*)d_in[5];
    const float*         gWK   = (const float*)d_in[6];
    const float*         gbK   = (const float*)d_in[7];
    const float*         gWV   = (const float*)d_in[8];
    const float*         gbV   = (const float*)d_in[9];
    const float*         gWO   = (const float*)d_in[10];
    const float*         gbO   = (const float*)d_in[11];
    float*               gOut  = (float*)d_out;

    dim3 grid(B_ / 4);
    dim3 block(128);
    mha_wmma_kernel<<<grid, block, SMEM_BYTES, stream>>>(
        gQ, gK, gV, gMask, gWQ, gbQ, gWK, gbK, gWV, gbV, gWO, gbO, gOut);
}
